// LSTMAutoencoder_19679540150780
// MI455X (gfx1250) — matrix operand order repaired and checked by reading, not compiled, not run on hardware
//
#include <hip/hip_runtime.h>
#include <hip/hip_fp16.h>

// LSTM autoencoder for MI455X (gfx1250, wave32, WMMA).
// B=512, T=512, D=128, H=64.
// 32 persistent workgroups x 256 threads (8 waves); each WG owns 16 batch rows
// for the full T=512 recurrence. Weights resident in LDS as f16; matmuls via
// v_wmma_f32_16x16x32_f16 with f32 accumulation; cell nonlinearities in f32.
// Encoder double-buffers x_t in LDS with register-staged prefetch of x_{t+1}
// (hides HBM latency behind the WMMA block) + global_prefetch of x_{t+2}.

typedef __attribute__((ext_vector_type(16))) _Float16 v16h;
typedef __attribute__((ext_vector_type(8)))  float    v8f;

#define B_SZ 512
#define T_SZ 512
#define D_SZ 128
#define H_SZ 64

__device__ __forceinline__ float sigm(float x) { return 1.0f / (1.0f + __expf(-x)); }
__device__ __forceinline__ float tanh_f(float x) {
    float e = __expf(2.0f * x);
    return (e - 1.0f) / (e + 1.0f);
}

// A-matrix fragment (16x32 f16), LDS row-major [16][ld], K-tile base k0.
// ISA 7.12.2: lanes 0-15 row M=lane, elems 0..7 = K k0..k0+7, elems 8..15 = K k0+16..k0+23;
// lanes 16-31 shift K base by +8.
__device__ __forceinline__ v16h load_frag_a(const _Float16* base, int ld, int k0) {
    const int lane = threadIdx.x & 31;
    const int row  = lane & 15;
    const _Float16* p = base + row * ld + k0 + ((lane >> 4) << 3);
    union { v16h v; uint4 q[2]; } u;
    u.q[0] = *(const uint4*)(p);
    u.q[1] = *(const uint4*)(p + 16);
    return u.v;
}

// B-matrix fragment (32x16 f16) from weights stored as WT[N][K] row-major
// (PyTorch weight [4Hc, K] layout IS this layout). b[e] = B[k0+8*(lane/16)+e+8*(e/8)][n0+lane%16].
__device__ __forceinline__ v16h load_frag_b(const _Float16* wt, int ld, int n0, int k0) {
    const int lane = threadIdx.x & 31;
    const _Float16* p = wt + (n0 + (lane & 15)) * ld + k0 + ((lane >> 4) << 3);
    union { v16h v; uint4 q[2]; } u;
    u.q[0] = *(const uint4*)(p);
    u.q[1] = *(const uint4*)(p + 16);
    return u.v;
}

// Initialize accumulator with broadcast bias[n0+col] (folds +b into the GEMM).
__device__ __forceinline__ v8f init_bias(const float* bias, int n0) {
    const int lane = threadIdx.x & 31;
    const float bv = bias[n0 + (lane & 15)];
    v8f c;
#pragma unroll
    for (int i = 0; i < 8; ++i) c[i] = bv;
    return c;
}

// C/D layout: VGPR v -> row = v + 8*(lane/16), col = lane%16.
__device__ __forceinline__ void store_acc(float* g, int ldg, int n0, v8f c) {
    const int lane  = threadIdx.x & 31;
    const int col   = n0 + (lane & 15);
    const int rbase = (lane >> 4) << 3;
#pragma unroll
    for (int v = 0; v < 8; ++v) g[(rbase + v) * ldg + col] = c[v];
}

// Convert 8 f32 -> 8 f16 and store as one 16B LDS write.
__device__ __forceinline__ void store8h(_Float16* p, float4 a, float4 b) {
    union { _Float16 h[8]; uint4 q; } u;
    u.h[0] = (_Float16)a.x; u.h[1] = (_Float16)a.y;
    u.h[2] = (_Float16)a.z; u.h[3] = (_Float16)a.w;
    u.h[4] = (_Float16)b.x; u.h[5] = (_Float16)b.y;
    u.h[6] = (_Float16)b.z; u.h[7] = (_Float16)b.w;
    *(uint4*)p = u.q;
}

// ---------------- Encoder: x[B,T,128] -> h_final[B,64] ----------------
__global__ __launch_bounds__(256) void lstm_encoder(const float* __restrict__ x,
                                                    const float* __restrict__ Wih,   // [256][128]
                                                    const float* __restrict__ Whh,   // [256][64]
                                                    const float* __restrict__ bias,  // [256]
                                                    float* __restrict__ h_out) {     // [B][64]
    extern __shared__ __align__(16) char smem[];
    _Float16* sWih = (_Float16*)smem;            // 256*128 f16
    _Float16* sWhh = sWih + 256 * 128;           // 256*64
    _Float16* sXA  = sWhh + 256 * 64;            // [2][16*128] double-buffered x_t, A layout
    _Float16* sHA  = sXA + 2 * 16 * 128;         // 16*64   (h state, A layout)
    float* sB = (float*)(sHA + 16 * 64);         // 256
    float* sC = sB + 256;                        // 16*64   (c state f32)
    float* sG = sC + 16 * 64;                    // 16*256  (gate staging f32)

    const int tid  = threadIdx.x;
    const int wave = tid >> 5;
    const int b0   = blockIdx.x * 16;
    const int lrow = tid >> 4;                   // 0..15  (x staging row)
    const int lcol = (tid & 15) * 8;             // 0..120 (x staging col, 8 floats/thread)

    for (int i = tid; i < 256 * 128; i += 256) sWih[i] = (_Float16)Wih[i];
    for (int i = tid; i < 256 * 64;  i += 256) sWhh[i] = (_Float16)Whh[i];
    if (tid < 256) sB[tid] = bias[tid];
    for (int i = tid; i < 16 * 64; i += 256) { sC[i] = 0.0f; sHA[i] = (_Float16)0.0f; }
    {   // stage x_0 into buffer 0
        const float* px = &x[((size_t)(b0 + lrow) * T_SZ + 0) * D_SZ + lcol];
        float4 f0 = *(const float4*)(px);
        float4 f1 = *(const float4*)(px + 4);
        store8h(&sXA[lrow * 128 + lcol], f0, f1);
    }
    __syncthreads();

    for (int t = 0; t < T_SZ; ++t) {
        const int cur = t & 1, nxt = cur ^ 1;
        const bool have_next = (t + 1 < T_SZ);

        // Issue global loads for x_{t+1} NOW; the s_wait lands after the WMMA block.
        float4 f0{}, f1{};
        if (have_next) {
            const float* px = &x[((size_t)(b0 + lrow) * T_SZ + (t + 1)) * D_SZ + lcol];
            f0 = *(const float4*)(px);
            f1 = *(const float4*)(px + 4);
        }
        if (t + 2 < T_SZ)   // warm L2 (192 MB; whole x is 134 MB) -> global_prefetch_b8
            __builtin_prefetch(&x[((size_t)(b0 + lrow) * T_SZ + (t + 2)) * D_SZ + lcol], 0, 1);

        // gates[16][256] = [x_t | h] * [Wih ; Whh]^T + bias   (fused K = 128 + 64)
        const _Float16* xbuf = sXA + cur * (16 * 128);
        v16h a[6];
#pragma unroll
        for (int kt = 0; kt < 4; ++kt) a[kt] = load_frag_a(xbuf, 128, kt * 32);
        a[4] = load_frag_a(sHA, 64, 0);
        a[5] = load_frag_a(sHA, 64, 32);

#pragma unroll
        for (int nt = 0; nt < 2; ++nt) {              // 16 N-tiles / 8 waves
            const int n0 = (wave * 2 + nt) * 16;
            v8f acc = init_bias(sB, n0);
#pragma unroll
            for (int kt = 0; kt < 4; ++kt) {
                v16h bf = load_frag_b(sWih, 128, n0, kt * 32);
                acc = __builtin_amdgcn_wmma_f32_16x16x32_f16(false, a[kt], false, bf,
                                                             (short)0, acc, false, false);
            }
#pragma unroll
            for (int kt = 0; kt < 2; ++kt) {
                v16h bf = load_frag_b(sWhh, 64, n0, kt * 32);
                acc = __builtin_amdgcn_wmma_f32_16x16x32_f16(false, a[4 + kt], false, bf,
                                                             (short)0, acc, false, false);
            }
            store_acc(sG, 256, n0, acc);
        }

        // Park x_{t+1} into the other buffer (nobody reads it this step).
        if (have_next)
            store8h(&sXA[nxt * (16 * 128) + lrow * 128 + lcol], f0, f1);
        __syncthreads();

        // LSTM cell: c = sig(f)*c + sig(i)*tanh(g); h = sig(o)*tanh(c)
        for (int i = tid; i < 16 * 64; i += 256) {
            const int r = i >> 6, j = i & 63;
            const float ig = sigm(sG[r * 256 + j]);
            const float fg = sigm(sG[r * 256 + 64 + j]);
            const float gg = tanh_f(sG[r * 256 + 128 + j]);
            const float og = sigm(sG[r * 256 + 192 + j]);
            const float c  = fg * sC[i] + ig * gg;
            sC[i] = c;
            const float h = og * tanh_f(c);
            sHA[i] = (_Float16)h;
            if (t == T_SZ - 1) h_out[(size_t)(b0 + r) * H_SZ + j] = h;
        }
        __syncthreads();
    }
}

// ---------------- Decoder: h_enc[B,64] (tiled over T) -> out[B,T,128] ----------------
__global__ __launch_bounds__(256) void lstm_decoder(const float* __restrict__ h_enc,  // [B][64]
                                                    const float* __restrict__ Wih,    // [512][64]
                                                    const float* __restrict__ Whh,    // [512][128]
                                                    const float* __restrict__ bias,   // [512]
                                                    float* __restrict__ out) {        // [B][T][128]
    extern __shared__ __align__(16) char smem[];
    _Float16* sWih = (_Float16*)smem;            // 512*64 f16
    _Float16* sWhh = sWih + 512 * 64;            // 512*128
    _Float16* sHE  = sWhh + 512 * 128;           // 16*64  (encoder hidden, A layout)
    _Float16* sHA  = sHE + 16 * 64;              // 16*128 (h state, A layout)
    float* sB = (float*)(sHA + 16 * 128);        // 512
    float* sC = sB + 512;                        // 16*128
    float* sG = sC + 16 * 128;                   // 16*512

    const int tid  = threadIdx.x;
    const int wave = tid >> 5;
    const int b0   = blockIdx.x * 16;

    for (int i = tid; i < 512 * 64;  i += 256) sWih[i] = (_Float16)Wih[i];
    for (int i = tid; i < 512 * 128; i += 256) sWhh[i] = (_Float16)Whh[i];
    for (int i = tid; i < 512; i += 256) sB[i] = bias[i];
    for (int i = tid; i < 16 * 64; i += 256)
        sHE[i] = (_Float16)h_enc[(size_t)(b0 + (i >> 6)) * H_SZ + (i & 63)];
    for (int i = tid; i < 16 * 128; i += 256) { sC[i] = 0.0f; sHA[i] = (_Float16)0.0f; }
    __syncthreads();

    // Input projection is constant over t (decoder input = tiled h_enc):
    // xp[16][512] = h_enc * Wih^T + bias, kept in accumulator registers.
    v8f xp[4];
    {
        const v16h a0 = load_frag_a(sHE, 64, 0);
        const v16h a1 = load_frag_a(sHE, 64, 32);
#pragma unroll
        for (int nt = 0; nt < 4; ++nt) {              // 32 N-tiles / 8 waves
            const int n0 = (wave * 4 + nt) * 16;
            v8f acc = init_bias(sB, n0);
            v16h bf = load_frag_b(sWih, 64, n0, 0);
            acc = __builtin_amdgcn_wmma_f32_16x16x32_f16(false, a0, false, bf,
                                                         (short)0, acc, false, false);
            bf = load_frag_b(sWih, 64, n0, 32);
            acc = __builtin_amdgcn_wmma_f32_16x16x32_f16(false, a1, false, bf,
                                                         (short)0, acc, false, false);
            xp[nt] = acc;
        }
    }

    for (int t = 0; t < T_SZ; ++t) {
        v16h a[4];
#pragma unroll
        for (int kt = 0; kt < 4; ++kt) a[kt] = load_frag_a(sHA, 128, kt * 32);

#pragma unroll
        for (int nt = 0; nt < 4; ++nt) {
            const int n0 = (wave * 4 + nt) * 16;
            v8f acc = xp[nt];                          // xp already holds bias + input proj
#pragma unroll
            for (int kt = 0; kt < 4; ++kt) {
                v16h bf = load_frag_b(sWhh, 128, n0, kt * 32);
                acc = __builtin_amdgcn_wmma_f32_16x16x32_f16(false, a[kt], false, bf,
                                                             (short)0, acc, false, false);
            }
            store_acc(sG, 512, n0, acc);
        }
        __syncthreads();

        for (int i = tid; i < 16 * 128; i += 256) {
            const int r = i >> 7, j = i & 127;
            const float ig = sigm(sG[r * 512 + j]);
            const float fg = sigm(sG[r * 512 + 128 + j]);
            const float gg = tanh_f(sG[r * 512 + 256 + j]);
            const float og = sigm(sG[r * 512 + 384 + j]);
            const float c  = fg * sC[i] + ig * gg;
            sC[i] = c;
            const float h = og * tanh_f(c);
            sHA[i] = (_Float16)h;
            out[((size_t)(b0 + r) * T_SZ + t) * D_SZ + j] = h;   // coalesced over j
        }
        __syncthreads();
    }
}

extern "C" void kernel_launch(void* const* d_in, const int* in_sizes, int n_in,
                              void* d_out, int out_size, void* d_ws, size_t ws_size,
                              hipStream_t stream) {
    (void)in_sizes; (void)n_in; (void)out_size; (void)ws_size;

    const float* x     = (const float*)d_in[0];
    const float* Wih_e = (const float*)d_in[1];
    const float* Whh_e = (const float*)d_in[2];
    const float* b_e   = (const float*)d_in[3];
    const float* Wih_d = (const float*)d_in[4];
    const float* Whh_d = (const float*)d_in[5];
    const float* b_d   = (const float*)d_in[6];
    float* outp = (float*)d_out;
    float* h_ws = (float*)d_ws;   // [B][64] encoder final hidden

    const int enc_smem = (256 * 128 + 256 * 64 + 2 * 16 * 128 + 16 * 64) * 2 +
                         (256 + 16 * 64 + 16 * 256) * 4;                   // ~127 KB
    const int dec_smem = (512 * 64 + 512 * 128 + 16 * 64 + 16 * 128) * 2 +
                         (512 + 16 * 128 + 16 * 512) * 4;                  // ~240 KB

    // gfx1250 WGP has 320 KB LDS; raise the dynamic-LDS cap (idempotent, capture-safe).
    (void)hipFuncSetAttribute(reinterpret_cast<const void*>(lstm_encoder),
                              hipFuncAttributeMaxDynamicSharedMemorySize, enc_smem);
    (void)hipFuncSetAttribute(reinterpret_cast<const void*>(lstm_decoder),
                              hipFuncAttributeMaxDynamicSharedMemorySize, dec_smem);

    lstm_encoder<<<B_SZ / 16, 256, enc_smem, stream>>>(x, Wih_e, Whh_e, b_e, h_ws);
    lstm_decoder<<<B_SZ / 16, 256, dec_smem, stream>>>(h_ws, Wih_d, Whh_d, b_d, outp);
}
